// CompressionAttn_3513283248262
// MI455X (gfx1250) — matrix operand order repaired and checked by reading, not compiled, not run on hardware
//
#include <hip/hip_runtime.h>

// ---------------- problem constants (match reference) ----------------
constexpr int BS     = 2;
constexpr int S      = 4096;
constexpr int HQ     = 32;
constexpr int HKV    = 2;
constexpr int D      = 128;
constexpr int STRIDE = 16;
constexpr int BLOCK  = 32;
constexpr int NB     = (S - BLOCK) / STRIDE + 1;   // 255
constexpr int NBP    = 256;                        // padded
constexpr int G      = HQ / HKV;                   // 16
constexpr int KD     = BLOCK * D;                  // 4096 (compress K-dim)

// ---------------- WMMA types ----------------
typedef __attribute__((ext_vector_type(16))) __bf16 v16bf;
typedef __attribute__((ext_vector_type(8)))  __bf16 v8bf;
typedef __attribute__((ext_vector_type(8)))  float  v8f;

static __device__ __forceinline__ v8f wmma_bf16(v16bf a, v16bf b, v8f c) {
  return __builtin_amdgcn_wmma_f32_16x16x32_bf16(
      /*neg_a=*/false, a, /*neg_b=*/false, b,
      /*c_mod=*/(short)0, c, /*reuse_a=*/false, /*reuse_b=*/false);
}

// ---------------- prep: transpose weight (BLOCK*D, D) f32 -> (D, BLOCK*D) bf16 ----
__global__ __launch_bounds__(256)
void nsa_transpose_w(const float* __restrict__ w, __bf16* __restrict__ wT) {
  int idx = blockIdx.x * 256 + threadIdx.x;          // over KD*D elements
  if (idx < KD * D) {
    int c = idx / D, d = idx % D;                    // read of w is coalesced
    wT[(size_t)d * KD + c] = (__bf16)w[idx];
  }
}

// ---------------- compress: (kc or vcT) = ((x blocks)+pe) @ w ----------------
// grid: (BS*HKV, NBP/16) ; block: 256 (8 waves, each owns a 16-wide slice of D)
// transposed==0 : out[b][h][nb][d]   (scores B operand)
// transposed==1 : out[b][h][d][nb]   (PV     B operand)
__global__ __launch_bounds__(256)
void nsa_compress(const float* __restrict__ x,     // (BS*S, HKV, D)
                  const __bf16* __restrict__ wT,   // (D, KD) bf16
                  const float* __restrict__ pe,    // (BLOCK, D)
                  __bf16* __restrict__ outc,
                  int transposed) {
  const int bh   = blockIdx.x;
  const int b    = bh / HKV, h = bh % HKV;
  const int n0   = blockIdx.y * 16;                 // nb tile base
  const int wave = threadIdx.x >> 5;
  const int lane = threadIdx.x & 31;
  const int d0   = wave * 16;
  const int hf   = lane >> 4;                       // lane half
  const int mr   = lane & 15;

  int nn = n0 + mr; if (nn > NB - 1) nn = NB - 1;   // clamp pad row (masked later)
  const int srow0 = nn * STRIDE;

  const __bf16* wrow = wT + (size_t)(d0 + mr) * KD; // B: lane = column n
  v8f acc = {};

  for (int kb = 0; kb < KD; kb += 32) {
    v16bf a;
#pragma unroll
    for (int run = 0; run < 2; ++run) {
      const int c  = kb + 16 * hf + run * 8;        // flat column in (j,e)
      const int j  = c >> 7;                        // block row
      const int e0 = c & (D - 1);
      const float* xs = x + ((size_t)(b * S + srow0 + j) * HKV + h) * D + e0;
      const float* ps = pe + j * D + e0;
#pragma unroll
      for (int e = 0; e < 8; ++e)
        a[run * 8 + e] = (__bf16)(xs[e] + ps[e]);
    }
    v16bf bm = *(const v16bf*)(wrow + kb + 16 * hf);
    acc = wmma_bf16(a, bm, acc);
  }

  if (!transposed) {
#pragma unroll
    for (int r = 0; r < 8; ++r) {
      int nb = n0 + r + 8 * hf;
      outc[((size_t)(b * HKV + h) * NBP + nb) * D + d0 + mr] = (__bf16)acc[r];
    }
  } else {
    __bf16* dst = outc + ((size_t)(b * HKV + h) * D + d0 + mr) * NBP + n0 + 8 * hf;
#pragma unroll
    for (int r = 0; r < 8; ++r) dst[r] = (__bf16)acc[r];
  }
}

// ---------------- attention: 4 waves per (b, hq, 16-query tile) ----------------
__global__ __launch_bounds__(128)
void nsa_attn(const float* __restrict__ q,      // (BS*S, HQ, D)
              const __bf16* __restrict__ kc,    // [b][h][NBP][D]
              const __bf16* __restrict__ vcT,   // [b][h][D][NBP]
              float* __restrict__ out,          // (BS,S,HQ,D)
              float* __restrict__ prob,         // (BS,HQ,S,NB)
              const float* __restrict__ scale_p,
              const unsigned char* __restrict__ causal_p) {
  __shared__ __align__(16) float  sc[16][NBP];    // scores -> probabilities (f32)
  __shared__ __align__(16) __bf16 pbf[16][NBP];   // probabilities (bf16, PV A operand)
  __shared__ __align__(16) float  otile[16][D];
  __shared__ float red[16][8];

  const int s0   = blockIdx.x * 16;
  const int hq   = blockIdx.y;
  const int b    = blockIdx.z;
  const int h    = hq / G;
  const int tid  = threadIdx.x;
  const int wave = tid >> 5;
  const int lane = tid & 31;
  const int hf   = lane >> 4;
  const int mr   = lane & 15;

  const float scale = *scale_p;
  const int causal  = (int)(*causal_p);

  const float*  qrow = q   + ((size_t)(b * S + s0 + mr) * HQ + hq) * D;
  const __bf16* kcb  = kc  + (size_t)(b * HKV + h) * NBP * D;
  const __bf16* vtb  = vcT + (size_t)(b * HKV + h) * D * NBP;

  // ---- hoisted Q fragments: A[m=mr][K], nt-invariant ----
  v16bf qa[4];
#pragma unroll
  for (int c = 0; c < 4; ++c) {
#pragma unroll
    for (int run = 0; run < 2; ++run) {
      const float* qs = qrow + c * 32 + 16 * hf + run * 8;
#pragma unroll
      for (int e = 0; e < 8; ++e) qa[c][run * 8 + e] = (__bf16)qs[e];
    }
  }

  // ---- scores: each wave computes 4 of the 16 nb-tiles ----
#pragma unroll
  for (int ww = 0; ww < 4; ++ww) {
    const int nt = wave * 4 + ww;
    v8f acc = {};
    const __bf16* krow = kcb + (size_t)(nt * 16 + mr) * D;  // B lane column = block
#pragma unroll
    for (int c = 0; c < 4; ++c) {
      v16bf bm = *(const v16bf*)(krow + c * 32 + 16 * hf);
      acc = wmma_bf16(qa[c], bm, acc);
    }
#pragma unroll
    for (int r = 0; r < 8; ++r)
      sc[r + 8 * hf][nt * 16 + mr] = acc[r] * scale;
  }
  __syncthreads();

  // ---- masked softmax over NB: 8 lanes per query row, 32 cols each ----
  {
    const int row = tid >> 3;
    const int seg = tid & 7;
    const int t   = s0 + row;
    int nvalid = causal ? ((t >= BLOCK - 1) ? ((t - (BLOCK - 1)) / STRIDE + 1) : 0) : NB;
    if (nvalid > NB) nvalid = NB;
    const int j0 = seg * 32, j1 = j0 + 32;

    float mx = -3.0e30f;
    for (int j = j0; j < j1; ++j)
      if (j < nvalid) mx = fmaxf(mx, sc[row][j]);
    red[row][seg] = mx;
    __syncthreads();
#pragma unroll
    for (int i2 = 0; i2 < 8; ++i2) mx = fmaxf(mx, red[row][i2]);
    __syncthreads();

    float sum = 0.f;
    for (int j = j0; j < j1; ++j) {
      float e = (j < nvalid) ? __expf(sc[row][j] - mx) : 0.f;
      sc[row][j] = e;
      sum += e;
    }
    red[row][seg] = sum;
    __syncthreads();
    sum = 0.f;
#pragma unroll
    for (int i2 = 0; i2 < 8; ++i2) sum += red[row][i2];
    const float inv = 1.0f / fmaxf(sum, 1e-20f);
    for (int j = j0; j < j1; ++j) {
      float p = sc[row][j] * inv;
      sc[row][j]  = p;            // f32 for attn_prob output
      pbf[row][j] = (__bf16)p;    // bf16 for PV A operand
    }
  }
  __syncthreads();

  // ---- write attn_prob[b][hq][s][nb] (coalesced across the 128 threads) ----
  {
    float* pbase = prob + ((size_t)(b * HQ + hq) * S + s0) * NB;
    for (int r = 0; r < 16; ++r)
      for (int j = tid; j < NB; j += 128)
        pbase[(size_t)r * NB + j] = sc[r][j];
  }

  // ---- hoisted P fragments from LDS (bf16), nt-invariant ----
  v16bf pa[8];
#pragma unroll
  for (int c = 0; c < 8; ++c) {
    union { v16bf v; v8bf h2[2]; } u;
    u.h2[0] = *(const v8bf*)&pbf[mr][c * 32 + 16 * hf];
    u.h2[1] = *(const v8bf*)&pbf[mr][c * 32 + 16 * hf + 8];
    pa[c] = u.v;
  }

  // ---- O_tile(16 x D) = P . Vc : each wave computes 2 of the 8 d-tiles ----
#pragma unroll
  for (int ww = 0; ww < 2; ++ww) {
    const int nt = wave * 2 + ww;
    v8f acc = {};
    const __bf16* vrow = vtb + (size_t)(nt * 16 + mr) * NBP; // B lane column = d
#pragma unroll
    for (int c = 0; c < 8; ++c) {
      v16bf bm = *(const v16bf*)(vrow + c * 32 + 16 * hf);
      acc = wmma_bf16(pa[c], bm, acc);
    }
#pragma unroll
    for (int r = 0; r < 8; ++r) otile[r + 8 * hf][nt * 16 + mr] = acc[r];
  }
  __syncthreads();

  // ---- coalesced O store: each wave streams whole 128-float rows ----
#pragma unroll
  for (int rr = 0; rr < 4; ++rr) {
    const int r = rr * 4 + wave;
    float4 vv = *(const float4*)&otile[r][lane * 4];
    *(float4*)(out + ((size_t)(b * S + s0 + r) * HQ + hq) * D + lane * 4) = vv;
  }
}

// ---------------- launcher ----------------
extern "C" void kernel_launch(void* const* d_in, const int* in_sizes, int n_in,
                              void* d_out, int out_size, void* d_ws, size_t ws_size,
                              hipStream_t stream) {
  const float* q    = (const float*)d_in[0];
  const float* k    = (const float*)d_in[1];
  const float* v    = (const float*)d_in[2];
  // d_in[3] = cu_seqlens_k (unused: fixed BS*S layout)
  const float* wk   = (const float*)d_in[4];
  const float* wv   = (const float*)d_in[5];
  const float* pe_k = (const float*)d_in[6];
  const float* pe_v = (const float*)d_in[7];
  const unsigned char* causal = (const unsigned char*)d_in[8];
  const float* scale = (const float*)d_in[9];

  float* out  = (float*)d_out;
  float* prob = out + (size_t)BS * S * HQ * D;

  // workspace carving (≈2.6 MB total)
  char* ws = (char*)d_ws;
  __bf16* wkT = (__bf16*)ws; ws += (size_t)D * KD * sizeof(__bf16);
  __bf16* wvT = (__bf16*)ws; ws += (size_t)D * KD * sizeof(__bf16);
  __bf16* kc  = (__bf16*)ws; ws += (size_t)BS * HKV * NBP * D * sizeof(__bf16);
  __bf16* vcT = (__bf16*)ws; ws += (size_t)BS * HKV * D * NBP * sizeof(__bf16);

  nsa_transpose_w<<<dim3((KD * D + 255) / 256), 256, 0, stream>>>(wk, wkT);
  nsa_transpose_w<<<dim3((KD * D + 255) / 256), 256, 0, stream>>>(wv, wvT);

  nsa_compress<<<dim3(BS * HKV, NBP / 16), 256, 0, stream>>>(k, wkT, pe_k, kc, 0);
  nsa_compress<<<dim3(BS * HKV, NBP / 16), 256, 0, stream>>>(v, wvT, pe_v, vcT, 1);

  nsa_attn<<<dim3(S / 16, HQ, BS), 128, 0, stream>>>(q, kc, vcT, out, prob, scale, causal);
}
